// Router_19421842113125
// MI455X (gfx1250) — matrix operand order repaired and checked by reading, not compiled, not run on hardware
//
#include <hip/hip_runtime.h>
#include <stdint.h>

// MoE router: logits GEMM (16384x64x2048) via v_wmma_f32_16x16x32_f16.
// K loop is a 2-stage software pipeline: A tiles staged one step ahead by the
// gfx1250 async engine (GLOBAL_LOAD_ASYNC_TO_LDS_B128 / ASYNCcnt, double
// buffered), B fragments prefetched one step ahead into a rotating register
// set (unroll x2, no copies). Then softmax-max/argmax per token and the
// capacity-limited one-hot via chunked scan.

typedef _Float16 v16h __attribute__((ext_vector_type(16)));
typedef float    v8f  __attribute__((ext_vector_type(8)));
typedef float    v4f  __attribute__((ext_vector_type(4)));
typedef int      v4i  __attribute__((ext_vector_type(4)));

#define B_DIM 4
#define S_DIM 4096
#define D_DIM 2048
#define E_DIM 64
#define CAP   128
#define NCHUNK 32
#define CHUNK  (S_DIM / NCHUNK)   // 128 tokens per chunk

#if defined(__gfx1250__) && \
    __has_builtin(__builtin_amdgcn_global_load_async_to_lds_b128) && \
    __has_builtin(__builtin_amdgcn_s_wait_asynccnt)
#define USE_ASYNC_LDS 1
#else
#define USE_ASYNC_LDS 0
#endif

#if USE_ASYNC_LDS
// Builtin signature (probe-confirmed via diagnostic): (AS1 int4*, AS3 int4*,
// int offset, int cpol).
#define ASYNC_B128(gp, lp)                                                \
    __builtin_amdgcn_global_load_async_to_lds_b128(                       \
        (__attribute__((address_space(1))) v4i*)(gp),                     \
        (__attribute__((address_space(3))) v4i*)(lp), 0, 0)
#endif

// ---------------------------------------------------------------------------
// Kernel 0: one-time W f32 -> f16 conversion (512 KB -> 256 KB, L2 resident)
// ---------------------------------------------------------------------------
__global__ void convert_w_kernel(const float* __restrict__ W,
                                 _Float16* __restrict__ Wh, int n) {
    int i = blockIdx.x * blockDim.x + threadIdx.x;
    if (i < n) Wh[i] = (_Float16)W[i];
}

// Convert a 16x32 f32 LDS tile row-run into the WMMA A fragment.
// lane -> m=l&15, hi=l>>4; half i: K=16hi+i.
__device__ __forceinline__ v16h cvt_a_frag(const float* __restrict__ Abuf,
                                           int m, int hi) {
    v4f a0 = *(const v4f*)(Abuf + m * 32 + 16 * hi);
    v4f a1 = *(const v4f*)(Abuf + m * 32 + 16 * hi + 4);
    v4f a2 = *(const v4f*)(Abuf + m * 32 + 16 * hi + 8);
    v4f a3 = *(const v4f*)(Abuf + m * 32 + 16 * hi + 12);
    v16h a;
#pragma unroll
    for (int i = 0; i < 4; ++i) {
        a[i]      = (_Float16)a0[i];
        a[4 + i]  = (_Float16)a1[i];
        a[8 + i]  = (_Float16)a2[i];
        a[12 + i] = (_Float16)a3[i];
    }
    return a;
}

// ---------------------------------------------------------------------------
// Kernel 1: router GEMM + softmax-max + argmax.
// One wave per 16-token tile; 4 x 16x16 f32 accumulators cover all 64 experts.
// B fragment (32x16 f16): lane l -> n=l&15, hi=l>>4, half i: K = kb + 16*hi + i
// C/D (16x16 f32): component r -> token M = r + 8*hi, expert N = (l&15)+16g.
// ---------------------------------------------------------------------------
__global__ void __launch_bounds__(256)
router_gemm_kernel(const float* __restrict__ hidden,
                   const _Float16* __restrict__ Wh,
                   const float* __restrict__ bias,
                   float* __restrict__ maxp,     // [B*S]
                   int* __restrict__ eidx)       // [B*S]
{
    const int lane = threadIdx.x & 31;
    const int wave = threadIdx.x >> 5;
    const int tile = blockIdx.x * 8 + wave;      // 128 blocks * 8 waves = 1024 tiles
    const int tokenBase = tile * 16;
    const int m  = lane & 15;
    const int hi = lane >> 4;

    v8f acc[4];
#pragma unroll
    for (int g = 0; g < 4; ++g)
#pragma unroll
        for (int r = 0; r < 8; ++r) acc[g][r] = 0.0f;

    // per-lane B base pointers (f16 rows, L2-resident)
    const _Float16* Bbase[4];
#pragma unroll
    for (int g = 0; g < 4; ++g)
        Bbase[g] = Wh + (size_t)(g * 16 + m) * D_DIM + 16 * hi;

#define LOAD_B(kb, bb)                                                       \
    do {                                                                     \
        _Pragma("unroll") for (int g = 0; g < 4; ++g)                        \
            (bb)[g] = *(const v16h*)(Bbase[g] + (kb));                       \
    } while (0)

#define WMMA4(a, bb)                                                         \
    do {                                                                     \
        _Pragma("unroll") for (int g = 0; g < 4; ++g)                        \
            acc[g] = __builtin_amdgcn_wmma_f32_16x16x32_f16(                 \
                false, (a), false, (bb)[g], (short)0, acc[g], false, false); \
    } while (0)

#if USE_ASYNC_LDS
    __shared__ float smemA[8][2][16 * 32];       // [wave][buf][row*32+k], 32 KB

    // per-lane async source pointers + per-buffer LDS destinations (hoisted)
    const float* gsrc[4];
    float* ld0[4];
    float* ld1[4];
#pragma unroll
    for (int j = 0; j < 4; ++j) {
        int c = j * 32 + lane;                   // chunk id 0..127
        int r = c >> 3;                          // tile row 0..15
        int p = c & 7;                           // 16B part 0..7
        gsrc[j] = hidden + (size_t)(tokenBase + r) * D_DIM + p * 4;
        ld0[j]  = &smemA[wave][0][r * 32 + p * 4];
        ld1[j]  = &smemA[wave][1][r * 32 + p * 4];
    }

#define ISSUE_A(kb, ld)                                                      \
    do {                                                                     \
        _Pragma("unroll") for (int j = 0; j < 4; ++j)                        \
            ASYNC_B128(gsrc[j] + (kb), (ld)[j]);                             \
    } while (0)

    // ---- prologue: tile 0 in flight, B(0) in flight --------------------
    ISSUE_A(0, ld0);
    v16h bb0[4], bb1[4];
    LOAD_B(0, bb0);

    // ---- steady state: 31 unrolled pairs (branch-free body) ------------
    int kb = 0;
    for (; kb + 64 < D_DIM; kb += 64) {
        ISSUE_A(kb + 32, ld1);                     // A one step ahead
        __builtin_amdgcn_s_wait_asynccnt(4);       // tile(kb) landed
        v16h a0 = cvt_a_frag(&smemA[wave][0][0], m, hi);
        LOAD_B(kb + 32, bb1);                      // B one step ahead
        WMMA4(a0, bb0);

        ISSUE_A(kb + 64, ld0);
        __builtin_amdgcn_s_wait_asynccnt(4);       // tile(kb+32) landed
        v16h a1 = cvt_a_frag(&smemA[wave][1][0], m, hi);
        LOAD_B(kb + 64, bb0);
        WMMA4(a1, bb1);
    }

    // ---- epilogue: kb == D_DIM-64, no prefetch past the end ------------
    ISSUE_A(kb + 32, ld1);
    __builtin_amdgcn_s_wait_asynccnt(4);
    v16h a0 = cvt_a_frag(&smemA[wave][0][0], m, hi);
    LOAD_B(kb + 32, bb1);
    WMMA4(a0, bb0);

    __builtin_amdgcn_s_wait_asynccnt(0);
    v16h a1 = cvt_a_frag(&smemA[wave][1][0], m, hi);
    WMMA4(a1, bb1);
#undef ISSUE_A
#else
    const float* __restrict__ Arow = hidden + (size_t)(tokenBase + m) * D_DIM;

    v4f a0 = *(const v4f*)(Arow + 16 * hi);
    v4f b0 = *(const v4f*)(Arow + 16 * hi + 4);
    v4f c0 = *(const v4f*)(Arow + 16 * hi + 8);
    v4f d0 = *(const v4f*)(Arow + 16 * hi + 12);

    for (int kb = 0; kb < D_DIM; kb += 32) {
        v16h a;
#pragma unroll
        for (int i = 0; i < 4; ++i) {
            a[i]      = (_Float16)a0[i];
            a[4 + i]  = (_Float16)b0[i];
            a[8 + i]  = (_Float16)c0[i];
            a[12 + i] = (_Float16)d0[i];
        }
        v16h bb[4];
        LOAD_B(kb, bb);
        if (kb + 32 < D_DIM) {
            const float* An = Arow + kb + 32;
            a0 = *(const v4f*)(An + 16 * hi);
            b0 = *(const v4f*)(An + 16 * hi + 4);
            c0 = *(const v4f*)(An + 16 * hi + 8);
            d0 = *(const v4f*)(An + 16 * hi + 12);
        }
        WMMA4(a, bb);
    }
#endif
#undef LOAD_B
#undef WMMA4

    // bias add: acc[g][r] is logits[token=r+8hi][e=16g+m]
#pragma unroll
    for (int g = 0; g < 4; ++g) {
        float be = bias[g * 16 + m];
#pragma unroll
        for (int r = 0; r < 8; ++r) acc[g][r] += be;
    }

    // per-token (per component r) max / argmax / sum-exp across the 16-lane half
#pragma unroll
    for (int r = 0; r < 8; ++r) {
        float mx = acc[0][r];
        int   mi = m;
#pragma unroll
        for (int g = 1; g < 4; ++g) {
            float v = acc[g][r];
            int   e = g * 16 + m;
            if (v > mx) { mx = v; mi = e; }
        }
#pragma unroll
        for (int msk = 1; msk < 16; msk <<= 1) {
            float om = __shfl_xor(mx, msk, 32);
            int   oi = __shfl_xor(mi, msk, 32);
            if (om > mx || (om == mx && oi < mi)) { mx = om; mi = oi; }
        }
        float s = 0.0f;
#pragma unroll
        for (int g = 0; g < 4; ++g) s += __expf(acc[g][r] - mx);
#pragma unroll
        for (int msk = 1; msk < 16; msk <<= 1) s += __shfl_xor(s, msk, 32);

        if (m == 0) {
            int token = tokenBase + r + 8 * hi;
            maxp[token] = 1.0f / s;   // max softmax prob = 1/sum(exp(l-lmax))
            eidx[token] = mi;
        }
    }
}

// ---------------------------------------------------------------------------
// Kernel 2a: per-chunk expert histograms. grid (NCHUNK, B), 64 threads (=experts)
// ---------------------------------------------------------------------------
__global__ void __launch_bounds__(64)
hist_kernel(const int* __restrict__ eidx, int* __restrict__ hist) {
    const int chunk = blockIdx.x;
    const int b     = blockIdx.y;
    const int e     = threadIdx.x;
    __shared__ int sidx[CHUNK];
    const int base = b * S_DIM + chunk * CHUNK;
    sidx[e]      = eidx[base + e];
    sidx[e + 64] = eidx[base + e + 64];
    __syncthreads();
    int cnt = 0;
    for (int s = 0; s < CHUNK; ++s) cnt += (sidx[s] == e) ? 1 : 0;
    hist[(b * NCHUNK + chunk) * E_DIM + e] = cnt;
}

// ---------------------------------------------------------------------------
// Kernel 2b: replay chunk with prefix offset; write capacity-masked one-hot.
// Lanes = experts -> each token's 64 output floats written coalesced.
// ---------------------------------------------------------------------------
__global__ void __launch_bounds__(64)
capacity_kernel(const int* __restrict__ eidx, const int* __restrict__ hist,
                float* __restrict__ out) {
    const int chunk = blockIdx.x;
    const int b     = blockIdx.y;
    const int e     = threadIdx.x;
    __shared__ int sidx[CHUNK];
    const int base = b * S_DIM + chunk * CHUNK;
    sidx[e]      = eidx[base + e];
    sidx[e + 64] = eidx[base + e + 64];
    __syncthreads();

    int cnt = 0;
    for (int c = 0; c < chunk; ++c) cnt += hist[(b * NCHUNK + c) * E_DIM + e];

    for (int s = 0; s < CHUNK; ++s) {
        bool match = (sidx[s] == e);
        cnt += match ? 1 : 0;
        out[(size_t)(base + s) * E_DIM + e] = (match && cnt <= CAP) ? 1.0f : 0.0f;
    }
}

// ---------------------------------------------------------------------------
extern "C" void kernel_launch(void* const* d_in, const int* in_sizes, int n_in,
                              void* d_out, int out_size, void* d_ws, size_t ws_size,
                              hipStream_t stream) {
    const float* hidden = (const float*)d_in[0];   // [B,S,D] f32
    const float* W      = (const float*)d_in[1];   // [E,D]   f32
    const float* bias   = (const float*)d_in[2];   // [E]     f32

    float* out  = (float*)d_out;                                   // expert_indices [B*S*E]
    float* maxp = out + (size_t)B_DIM * S_DIM * E_DIM;             // max_probs      [B*S]

    // workspace: Wh f16 (256KB) | eidx int (64KB) | hist int (32KB)
    char* ws = (char*)d_ws;
    _Float16* Wh  = (_Float16*)ws;
    int* eidx     = (int*)(ws + (size_t)E_DIM * D_DIM * sizeof(_Float16));
    int* hist     = (int*)(ws + (size_t)E_DIM * D_DIM * sizeof(_Float16)
                              + (size_t)B_DIM * S_DIM * sizeof(int));

    const int nW = E_DIM * D_DIM;
    convert_w_kernel<<<(nW + 255) / 256, 256, 0, stream>>>(W, Wh, nW);

    // 16384 tokens / 16 per wave = 1024 wave-tiles; 8 waves per block
    router_gemm_kernel<<<128, 256, 0, stream>>>(hidden, Wh, bias, maxp, eidx);

    dim3 g2(NCHUNK, B_DIM);
    hist_kernel<<<g2, 64, 0, stream>>>(eidx, hist);
    capacity_kernel<<<g2, 64, 0, stream>>>(eidx, hist, out);
}
